// BiLSTM_CRF_3169685864931
// MI455X (gfx1250) — compile-verified, matrix operand order checked by reading, not run on hardware
//
#include <hip/hip_runtime.h>
#include <hip/hip_bf16.h>
#include <stdint.h>

#define SEQ   8192
#define BH    768
#define HID   500
#define GDIM  2000      // 4*HID rows per direction
#define NPAD  4096      // padded combined gate dim (2*2000 -> 4096 for 128-wide N tiles)
#define NREAL 4000
#define TAGS  9
#define START_TAG 7
#define STOP_TAG  8
#define NEGV  (-10000.0f)

typedef unsigned short u16;
typedef unsigned int   u32;
typedef unsigned long long u64;
typedef __attribute__((ext_vector_type(16))) __bf16 v16bf;
typedef __attribute__((ext_vector_type(8)))  __bf16 v8bf;
typedef __attribute__((ext_vector_type(8)))  float  v8f;
typedef __attribute__((ext_vector_type(4)))  u32    uv4;
typedef __attribute__((ext_vector_type(4)))  u32    u32x4;
typedef __attribute__((ext_vector_type(8)))  int    i32x8;
typedef __attribute__((ext_vector_type(4)))  int    i32x4;

#if defined(__HIP_DEVICE_COMPILE__) && __has_builtin(__builtin_amdgcn_tensor_load_to_lds)
#define HAS_TDM 1
#else
#define HAS_TDM 0
#endif

union FragBF { v16bf v; v8bf h[2]; };

__device__ __forceinline__ u16 f2bf(float f) {
  u32 u = __builtin_bit_cast(u32, f);
  u32 r = u + 0x7FFFu + ((u >> 16) & 1u);
  return (u16)(r >> 16);
}
__device__ __forceinline__ float bf2f(u16 h) {
  return __builtin_bit_cast(float, (u32)h << 16);
}
__device__ __forceinline__ float sigm(float x) { return 1.0f / (1.0f + __expf(-x)); }

#if HAS_TDM
// Issue one TDM load of a 128-row x 32-col bf16 tile (row stride = tdim0 elems)
// into LDS at lds_off. Descriptor layout per CDNA5 ISA 8.3/8.4 (D# group0/1).
__device__ __forceinline__ void tdm_load_tile(u32 lds_off, const u16* gptr,
                                              u32 tdim0, u32 trows) {
  const u64 ga = (u64)(size_t)gptr;
  u32x4 g0 = { 1u,                                   // count=1 (valid user D#)
               lds_off,                               // lds_addr (bytes)
               (u32)ga,                               // global_addr[31:0]
               ((u32)(ga >> 32) & 0x01FFFFFFu) | (2u << 30) };  // addr[56:32] | type=2
  i32x8 g1 = { (int)(1u << 16),                       // data_size=1 -> 2 bytes
               (int)((tdim0 & 0xFFFFu) << 16),        // tensor_dim0[15:0]
               (int)(((trows & 0xFFFFu) << 16) | (tdim0 >> 16)),  // tdim0 hi | tensor_dim1 lo
               (int)((32u << 16) | (trows >> 16)),    // tensor_dim1 hi | tile_dim0=32
               (int)128,                              // tile_dim1=128, tile_dim2=0
               (int)tdim0,                            // tensor_dim0_stride[31:0]
               0, 0 };                                // stride hi / dim1_stride = 0
  i32x4 z4 = { 0, 0, 0, 0 };
#if __clang_major__ >= 23
  i32x8 z8 = { 0, 0, 0, 0, 0, 0, 0, 0 };
  __builtin_amdgcn_tensor_load_to_lds(g0, g1, z4, z4, z8, 0);
#else
  __builtin_amdgcn_tensor_load_to_lds(g0, g1, z4, z4, 0);
#endif
}
#endif

// ---------------------------------------------------------------- conversions
__global__ void k_cvt_x(const float* __restrict__ x, u16* __restrict__ xb, int n) {
  int i = blockIdx.x * blockDim.x + threadIdx.x;
  if (i < n) xb[i] = f2bf(x[i]);
}

__global__ void k_build_w(const float* __restrict__ wf, const float* __restrict__ wb,
                          u16* __restrict__ Wbf) {
  int i = blockIdx.x * blockDim.x + threadIdx.x;
  if (i >= NPAD * BH) return;
  int r = i / BH, c = i % BH;
  float v = 0.0f;
  if (r < GDIM)            v = wf[(size_t)r * BH + c];
  else if (r < 2 * GDIM)   v = wb[(size_t)(r - GDIM) * BH + c];
  Wbf[i] = f2bf(v);
}

__global__ void k_build_whh(const float* __restrict__ wf, const float* __restrict__ wb,
                            u16* __restrict__ Whh) {
  int i = blockIdx.x * blockDim.x + threadIdx.x;
  const int per = GDIM * HID;
  if (i >= 2 * per) return;
  float v = (i < per) ? wf[i] : wb[i - per];
  Whh[i] = f2bf(v);
}

__global__ void k_build_bias(const float* __restrict__ bif, const float* __restrict__ bhf,
                             const float* __restrict__ bib, const float* __restrict__ bhb,
                             float* __restrict__ biasP) {
  int i = blockIdx.x * blockDim.x + threadIdx.x;
  if (i >= NPAD) return;
  float v = 0.0f;
  if (i < GDIM)          v = bif[i] + bhf[i];
  else if (i < 2 * GDIM) v = bib[i - GDIM] + bhb[i - GDIM];
  biasP[i] = v;
}

__global__ void k_init(const float* __restrict__ h0, float* __restrict__ hbuf,
                       int* __restrict__ syncc) {
  int i = blockIdx.x * blockDim.x + threadIdx.x;
  if (i < 1024) {                       // parity-0 region: hbuf[0][dir][512]
    int dir = i >> 9, u = i & 511;
    hbuf[dir * 512 + u] = (u < HID) ? h0[dir * HID + u] : 0.0f;
  }
  if (i < 2) syncc[i] = 0;
}

// ------------------------------------------------- WMMA bf16 GEMM: P = X*W^T + bias
// block tile 128(M) x 128(N), K step 32; 8 waves (4x2), each wave 32x64 (2x4 tiles).
// TDM double-buffered staging: wave 0 issues tensor_load_to_lds for the next K
// tile pair while all waves run WMMA on the current pair.
__global__ __launch_bounds__(256)
void k_gemm_wmma(const u16* __restrict__ Xbf, const u16* __restrict__ Wbf,
                 const float* __restrict__ biasP, float* __restrict__ P) {
  __shared__ u16 As[2][128 * 32];   // A tiles, row-major, ld=32
  __shared__ u16 Bs[2][128 * 32];   // W rows (== B columns), ld=32 in K
  const int tid  = threadIdx.x;
  const int lane = tid & 31, wave = tid >> 5;
  const int wm = wave & 3, wn = wave >> 2;
  const int m0 = blockIdx.x * 128, n0 = blockIdx.y * 128;
  const int row = lane & 15;
  const int hi  = (lane >> 4) << 3;      // K-half select per ISA 16-bit layout

  v8f acc[2][4] = {};

#if HAS_TDM
  if (wave == 0) {
    tdm_load_tile((u32)(size_t)(void*)&As[0][0], Xbf + (size_t)m0 * BH, BH, SEQ);
    tdm_load_tile((u32)(size_t)(void*)&Bs[0][0], Wbf + (size_t)n0 * BH, BH, NPAD);
  }
#endif

  int buf = 0;
  for (int kb = 0; kb < BH; kb += 32, buf ^= 1) {
#if HAS_TDM
    if (wave == 0) __builtin_amdgcn_s_wait_tensorcnt((short)0);  // pair(kb) arrived
    __syncthreads();   // all waves: buf valid; buf^1 readers from kb-32 all done
    if (wave == 0 && kb + 32 < BH) {
      tdm_load_tile((u32)(size_t)(void*)&As[buf ^ 1][0],
                    Xbf + (size_t)m0 * BH + kb + 32, BH, SEQ);
      tdm_load_tile((u32)(size_t)(void*)&Bs[buf ^ 1][0],
                    Wbf + (size_t)n0 * BH + kb + 32, BH, NPAD);
    }
#else
    for (int i = tid; i < 512; i += 256) {          // 512 x 16B chunks per tile
      int r = i >> 2, c = (i & 3) << 3;
      *(uv4*)(As[buf] + r * 32 + c) = *(const uv4*)(Xbf + (size_t)(m0 + r) * BH + kb + c);
      *(uv4*)(Bs[buf] + r * 32 + c) = *(const uv4*)(Wbf + (size_t)(n0 + r) * BH + kb + c);
    }
    __syncthreads();
#endif

    FragBF a[2], b[4];
#pragma unroll
    for (int mi = 0; mi < 2; ++mi) {
      const u16* p = As[buf] + (wm * 32 + mi * 16 + row) * 32;
      a[mi].h[0] = *(const v8bf*)(p + hi);          // K 0..7   / 8..15
      a[mi].h[1] = *(const v8bf*)(p + 16 + hi);     // K 16..23 / 24..31
    }
#pragma unroll
    for (int ni = 0; ni < 4; ++ni) {
      const u16* p = Bs[buf] + (wn * 64 + ni * 16 + row) * 32;
      b[ni].h[0] = *(const v8bf*)(p + hi);
      b[ni].h[1] = *(const v8bf*)(p + 16 + hi);
    }
#pragma unroll
    for (int mi = 0; mi < 2; ++mi)
#pragma unroll
      for (int ni = 0; ni < 4; ++ni)
        acc[mi][ni] = __builtin_amdgcn_wmma_f32_16x16x32_bf16(
            false, a[mi].v, false, b[ni].v, (short)0, acc[mi][ni], false, false);

#if !HAS_TDM
    __syncthreads();
#endif
  }

  // C/D layout: lanes 0-15 -> M=r, N=lane; lanes 16-31 -> M=r+8, N=lane-16
#pragma unroll
  for (int mi = 0; mi < 2; ++mi) {
    const int mbase = m0 + wm * 32 + mi * 16 + (lane >> 4) * 8;
#pragma unroll
    for (int ni = 0; ni < 4; ++ni) {
      const int n = n0 + wn * 64 + ni * 16 + (lane & 15);
      if (n < NREAL) {
        const float bv = biasP[n];
#pragma unroll
        for (int r = 0; r < 8; ++r)
          P[(size_t)(mbase + r) * NPAD + n] = acc[mi][ni][r] + bv;
      }
    }
  }
}

// ---------------------------------------------- sequential biLSTM recurrence
// 10 blocks per direction, each owns 50 hidden units (200 w_hh rows). Per-step
// grid sync via monotonic counter; h exchanged through parity-buffered global.
#define NBLK 10
#define UPB  50

__global__ __launch_bounds__(256)
void k_lstm(const float* __restrict__ P, const u16* __restrict__ Whh,
            const float* __restrict__ c0, float* __restrict__ hbuf,
            float* __restrict__ lstm_out, int* __restrict__ syncc) {
  const int tid = threadIdx.x;
  const int dir = blockIdx.x / NBLK;
  const int blk = blockIdx.x % NBLK;
  __shared__ float hsh[512];
  __shared__ float gsh[4 * UPB];
  __shared__ float csh[UPB];
  if (tid < UPB) csh[tid] = c0[dir * HID + blk * UPB + tid];

  for (int s = 0; s < SEQ; ++s) {
    const int t = dir ? (SEQ - 1 - s) : s;
    const float* hp = hbuf + (size_t)((s & 1) * 2 + dir) * 512;
    for (int i = tid; i < HID; i += 256) hsh[i] = hp[i];
    __syncthreads();

    if (tid < 4 * UPB) {
      const int gate = tid / UPB, u = tid % UPB;
      const int rowd = gate * HID + blk * UPB + u;
      float sum = P[(size_t)t * NPAD + dir * GDIM + rowd];
      const u32* w2 = (const u32*)(Whh + ((size_t)dir * GDIM + rowd) * HID);
#pragma unroll 4
      for (int k = 0; k < HID / 2; ++k) {
        const u32 pk = w2[k];
        sum += bf2f((u16)pk) * hsh[2 * k] + bf2f((u16)(pk >> 16)) * hsh[2 * k + 1];
      }
      gsh[tid] = sum;
    }
    __syncthreads();

    if (tid < UPB) {
      const float ig = sigm(gsh[tid]);
      const float fg = sigm(gsh[UPB + tid]);
      const float gg = tanhf(gsh[2 * UPB + tid]);
      const float og = sigm(gsh[3 * UPB + tid]);
      const float c  = fg * csh[tid] + ig * gg;
      csh[tid] = c;
      const float h = og * tanhf(c);
      const int ug = blk * UPB + tid;
      lstm_out[(size_t)t * (2 * HID) + dir * HID + ug] = h;
      hbuf[(size_t)(((s + 1) & 1) * 2 + dir) * 512 + ug] = h;
    }
    __threadfence();
    __syncthreads();
    if (tid == 0) {
      atomicAdd(&syncc[dir], 1);
      volatile int* vs = (volatile int*)&syncc[dir];
      const int tgt = NBLK * (s + 1);
      while (*vs < tgt) { __builtin_amdgcn_s_sleep(1); }
    }
    __syncthreads();
    __threadfence();
  }
}

// --------------------------------------------------------- tag projection
__global__ void k_tagproj(const float* __restrict__ lstm_out,
                          const float* __restrict__ w_tag, const float* __restrict__ b_tag,
                          float* __restrict__ feats) {
  const int gid = blockIdx.x * blockDim.x + threadIdx.x;
  if (gid >= SEQ * TAGS) return;
  const int t = gid / TAGS, j = gid % TAGS;
  const float* lr = lstm_out + (size_t)t * (2 * HID);
  const float* wr = w_tag + (size_t)j * (2 * HID);
  float sum = b_tag[j];
  for (int k = 0; k < 2 * HID; ++k) sum += lr[k] * wr[k];
  feats[gid] = sum;
}

// ------------------------------------------------------------- Viterbi DP
#define VCH 512
__global__ __launch_bounds__(32)
void k_viterbi(const float* __restrict__ feats, const float* __restrict__ trans,
               int* __restrict__ bptr, float* __restrict__ out) {
  const int tid = threadIdx.x;
  __shared__ float tr[TAGS * TAGS];
  __shared__ float fv[TAGS];
  __shared__ float fch[VCH * TAGS];
  __shared__ int   bch[VCH * TAGS];
  __shared__ int   pch[VCH];
  __shared__ int   curTag;
  for (int i = tid; i < TAGS * TAGS; i += 32) tr[i] = trans[i];
  if (tid < TAGS) fv[tid] = (tid == START_TAG) ? 0.0f : NEGV;
  __syncthreads();

  for (int c = 0; c < SEQ / VCH; ++c) {
    const int base = c * VCH;
    for (int i = tid; i < VCH * TAGS; i += 32) fch[i] = feats[(size_t)base * TAGS + i];
    __syncthreads();
    for (int s = 0; s < VCH; ++s) {
      float nv = 0.0f;
      if (tid < TAGS) {
        float best = fv[0] + tr[tid * TAGS + 0];
        int bi = 0;
        for (int p = 1; p < TAGS; ++p) {
          const float v = fv[p] + tr[tid * TAGS + p];
          if (v > best) { best = v; bi = p; }        // first-max like jnp.argmax
        }
        nv = best + fch[s * TAGS + tid];
        bch[s * TAGS + tid] = bi;
      }
      __syncthreads();
      if (tid < TAGS) fv[tid] = nv;
      __syncthreads();
    }
    for (int i = tid; i < VCH * TAGS; i += 32) bptr[(size_t)base * TAGS + i] = bch[i];
    __syncthreads();
  }

  if (tid == 0) {
    float best = fv[0] + tr[STOP_TAG * TAGS + 0];
    int bi = 0;
    for (int j = 1; j < TAGS; ++j) {
      const float v = fv[j] + tr[STOP_TAG * TAGS + j];
      if (v > best) { best = v; bi = j; }
    }
    out[0] = best;
    curTag = bi;
  }
  __syncthreads();

  for (int c = SEQ / VCH - 1; c >= 0; --c) {
    const int base = c * VCH;
    for (int i = tid; i < VCH * TAGS; i += 32) bch[i] = bptr[(size_t)base * TAGS + i];
    __syncthreads();
    if (tid == 0) {
      int cur = curTag;
      for (int s = VCH - 1; s >= 0; --s) {
        pch[s] = cur;
        cur = bch[s * TAGS + cur];
      }
      curTag = cur;
    }
    __syncthreads();
    for (int i = tid; i < VCH; i += 32) out[1 + base + i] = (float)pch[i];
    __syncthreads();
  }
}

// ---------------------------------------------------------------- launch
extern "C" void kernel_launch(void* const* d_in, const int* in_sizes, int n_in,
                              void* d_out, int out_size, void* d_ws, size_t ws_size,
                              hipStream_t stream) {
  (void)in_sizes; (void)n_in; (void)out_size; (void)ws_size;
  const float* emb   = (const float*)d_in[0];
  const float* wihf  = (const float*)d_in[1];
  const float* whhf  = (const float*)d_in[2];
  const float* bihf  = (const float*)d_in[3];
  const float* bhhf  = (const float*)d_in[4];
  const float* wihb  = (const float*)d_in[5];
  const float* whhb  = (const float*)d_in[6];
  const float* bihb  = (const float*)d_in[7];
  const float* bhhb  = (const float*)d_in[8];
  const float* h0    = (const float*)d_in[9];
  const float* c0    = (const float*)d_in[10];
  const float* wtag  = (const float*)d_in[11];
  const float* btag  = (const float*)d_in[12];
  const float* trans = (const float*)d_in[13];
  float* out = (float*)d_out;

  char* w = (char*)d_ws;
  float* P     = (float*)w;  w += (size_t)SEQ * NPAD * 4;      // 134.2 MB  xproj (fwd|bwd, padded)
  u16* Xbf     = (u16*)w;    w += (size_t)SEQ * BH * 2;        // 12.6 MB
  u16* Wbf     = (u16*)w;    w += (size_t)NPAD * BH * 2;       // 6.3 MB
  float* biasP = (float*)w;  w += (size_t)NPAD * 4;
  u16* Whh     = (u16*)w;    w += (size_t)2 * GDIM * HID * 2;  // 4 MB (L2-resident)
  float* hbuf  = (float*)w;  w += (size_t)2 * 2 * 512 * 4;     // parity-buffered h
  int* syncc   = (int*)w;    w += 256;
  float* lstm  = (float*)w;  w += (size_t)SEQ * 2 * HID * 4;   // 32.8 MB
  float* feats = (float*)w;  w += (size_t)SEQ * TAGS * 4;
  int* bptr    = (int*)w;    w += (size_t)SEQ * TAGS * 4;

  k_init<<<4, 256, 0, stream>>>(h0, hbuf, syncc);
  {
    const int n = SEQ * BH;
    k_cvt_x<<<(n + 255) / 256, 256, 0, stream>>>(emb, Xbf, n);
  }
  k_build_w<<<(NPAD * BH + 255) / 256, 256, 0, stream>>>(wihf, wihb, Wbf);
  k_build_whh<<<(2 * GDIM * HID + 255) / 256, 256, 0, stream>>>(whhf, whhb, Whh);
  k_build_bias<<<(NPAD + 255) / 256, 256, 0, stream>>>(bihf, bhhf, bihb, bhhb, biasP);

  dim3 gg(SEQ / 128, NPAD / 128);                  // 64 x 32 blocks
  k_gemm_wmma<<<gg, 256, 0, stream>>>(Xbf, Wbf, biasP, P);

  k_lstm<<<2 * NBLK, 256, 0, stream>>>(P, Whh, c0, hbuf, lstm, syncc);
  k_tagproj<<<(SEQ * TAGS + 255) / 256, 256, 0, stream>>>(lstm, wtag, btag, feats);
  k_viterbi<<<1, 32, 0, stream>>>(feats, trans, bptr, out);
}
